// mofnet_GNN_14465449853401
// MI455X (gfx1250) — matrix operand order repaired and checked by reading, not compiled, not run on hardware
//
#include <hip/hip_runtime.h>
#include <hip/hip_bf16.h>

// ---------------------------------------------------------------------------
// MOFnet (GNN + hypernetwork) for gfx1250.
// Heavy GEMMs run on v_wmma_f32_16x16x32_bf16. LDS tiles are stored in a
// fragment-friendly layout (A row-major, B transposed, 80B padded stride) so
// each WMMA operand loads as ds_load_b128s. Staging is branch-free and
// vectorized (float4 global loads, packed bf16 LDS stores). The per-edge
// weight tensor `ew` and hypernetwork weights are materialized in bf16 to
// halve HBM traffic (23.3 TB/s roofline: ~1.0 GB streamed -> ~45us floor).
// Activations use branch-free tanh/sigmoid built on v_exp_f32 + v_rcp_f32.
// ---------------------------------------------------------------------------

typedef __attribute__((ext_vector_type(16))) __bf16 v16bf;
typedef __attribute__((ext_vector_type(8)))  __bf16 v8bf;
typedef __attribute__((ext_vector_type(4)))  __bf16 v4bf;
typedef __attribute__((ext_vector_type(2)))  __bf16 v2bf;
typedef __attribute__((ext_vector_type(8)))  float  v8f;

union BF16x16 { v16bf v; v8bf h[2]; };

static constexpr int BB  = 512;      // MOFs
static constexpr int GG  = 1536;     // graphs per GNN (3*B)
static constexpr int NNODE = 24576;  // nodes per GNN (G*16)
static constexpr int NEDGE = 24576;  // edges per GNN (G*16)

#define BN_RSQ 0.9999950000374997f   // 1/sqrt(1 + 1e-5)

__device__ __forceinline__ float fast_rcp(float x) {
  return __builtin_amdgcn_rcpf(x);   // single v_rcp_f32
}

// branch-free saturating tanh: copysign(1 - 2/(exp(2|x|)+1), x).
// exp->inf  =>  rcp(inf) -> 0  =>  t -> 1 : correct saturation, no branches.
__device__ __forceinline__ float fast_tanh(float x) {
  float e = __expf(2.0f * fabsf(x));
  float t = 1.0f - 2.0f * fast_rcp(e + 1.0f);
  return copysignf(t, x);
}

__device__ __forceinline__ float sigm(float x) {
  return fast_rcp(1.0f + __expf(-x));
}

// ---------------------------- WMMA GEMM ------------------------------------
// C[M,N] = act(A[M,K] @ B[K,N] + bias[N]); A,B f32 in HBM, bf16 in LDS.
// Block tile 128x128, 256 threads = 8 waves (4 along M x 2 along N), each
// wave a 32x64 strip = 2x4 16x16 f32 accumulators.
// REQUIREMENTS: M % 128 == 0, K % 32 == 0, A 16-byte aligned. N arbitrary.
template<int ACT, bool OBF16>
__global__ __launch_bounds__(256)
void gemm_k(const float* __restrict__ A, const float* __restrict__ B,
            const float* __restrict__ bias, void* __restrict__ Cv,
            int N, int K)
{
  // padded stride 40 bf16 = 80 B (multiple of 16 B -> b128-aligned chunks)
  __shared__ __bf16 sA [128][40];   // A tile, row-major [m][k]
  __shared__ __bf16 sBt[128][40];   // B tile, TRANSPOSED [n][k]
  const int t    = threadIdx.x;
  const int lane = t & 31;
  const int wid  = t >> 5;
  const int lh   = lane >> 4;       // lane half (0/1)
  const int l16  = lane & 15;
  const int waveM = wid & 3;        // 4 waves along M (32 rows each)
  const int waveN = wid >> 2;       // 2 waves along N (64 cols each)
  const int rowBase = blockIdx.y * 128;
  const int colBase = blockIdx.x * 128;

  // B staging: this thread owns one column, 16 K rows (branch-free guard)
  const int bn   = t & 127;                 // column within tile
  const int bkb  = (t >> 7) * 16;           // first K row of this thread
  const int gcTh = colBase + bn;
  const float bSel = (gcTh < N) ? 1.0f : 0.0f;
  const int gcCl = (gcTh < N) ? gcTh : (N - 1);

  v8f acc[2][4];
#pragma unroll
  for (int mi = 0; mi < 2; ++mi)
#pragma unroll
    for (int n = 0; n < 4; ++n)
#pragma unroll
      for (int v = 0; v < 8; ++v) acc[mi][n][v] = 0.0f;

  for (int k0 = 0; k0 < K; k0 += 32) {
    // ---- stage A tile (128x32): 4 x float4 loads, packed bf16 stores ----
#pragma unroll
    for (int i = 0; i < 4; ++i) {
      int e4 = i * 256 + t;
      int r  = e4 >> 3;               // row in tile
      int cg = (e4 & 7) * 4;          // first of 4 cols
      float4 v = *(const float4*)(A + (size_t)(rowBase + r) * K + k0 + cg);
      v4bf p;
      p[0] = (__bf16)v.x; p[1] = (__bf16)v.y;
      p[2] = (__bf16)v.z; p[3] = (__bf16)v.w;
      *(v4bf*)&sA[r][cg] = p;
    }
    // ---- stage B tile (32x128) transposed: 16 coalesced b32 loads ----
    {
      const float* bp = B + (size_t)(k0 + bkb) * N + gcCl;
      v8bf lo, hi;
#pragma unroll
      for (int j = 0; j < 8; ++j)  lo[j] = (__bf16)(bp[(size_t)j * N] * bSel);
#pragma unroll
      for (int j = 0; j < 8; ++j)  hi[j] = (__bf16)(bp[(size_t)(j + 8) * N] * bSel);
      *(v8bf*)&sBt[bn][bkb]     = lo;
      *(v8bf*)&sBt[bn][bkb + 8] = hi;
    }
    __syncthreads();

    // ---- fragments: ds_load_b128 pairs ----
    BF16x16 af[2];
#pragma unroll
    for (int mi = 0; mi < 2; ++mi) {
      const __bf16* row = &sA[waveM * 32 + mi * 16 + l16][0];
      af[mi].h[0] = *(const v8bf*)(row + lh * 8);        // K lh*8 .. +7
      af[mi].h[1] = *(const v8bf*)(row + 16 + lh * 8);   // K 16+lh*8 .. +7
    }
#pragma unroll
    for (int n = 0; n < 4; ++n) {
      const __bf16* col = &sBt[waveN * 64 + n * 16 + l16][0];
      BF16x16 bfg;
      bfg.h[0] = *(const v8bf*)(col + lh * 8);
      bfg.h[1] = *(const v8bf*)(col + 16 + lh * 8);
#pragma unroll
      for (int mi = 0; mi < 2; ++mi) {
        acc[mi][n] = __builtin_amdgcn_wmma_f32_16x16x32_bf16(
            false, af[mi].v, false, bfg.v, (short)0, acc[mi][n], false, false);
      }
    }
    __syncthreads();
  }

  float*  Cf = (float*)Cv;
  __bf16* Cb = (__bf16*)Cv;
#pragma unroll
  for (int mi = 0; mi < 2; ++mi) {
#pragma unroll
    for (int n = 0; n < 4; ++n) {
      int gc = colBase + waveN * 64 + n * 16 + l16;
      if (gc >= N) continue;
      float bv = bias[gc];
#pragma unroll
      for (int v = 0; v < 8; ++v) {
        int gr = rowBase + waveM * 32 + mi * 16 + v + lh * 8;  // C/D layout
        float val = acc[mi][n][v] + bv;
        if (ACT == 1) val = fmaxf(val, 0.0f);
        if (ACT == 2) val = fast_tanh(val);
        if (OBF16) Cb[(size_t)gr * N + gc] = (__bf16)val;
        else       Cf[(size_t)gr * N + gc] = val;
      }
    }
  }
}

// ------------------------- elementwise / graph kernels ---------------------
__global__ void k_zero(float* p, long n) {
  long i = (long)blockIdx.x * blockDim.x + threadIdx.x;
  if (i < n) p[i] = 0.0f;
}

// nf = relu(x(N,2) @ W(2,64) + b)
__global__ void k_embed(const float* __restrict__ x, const float* __restrict__ W,
                        const float* __restrict__ b, float* __restrict__ nf, int N) {
  int idx = blockIdx.x * blockDim.x + threadIdx.x;
  if (idx >= N * 64) return;
  int n = idx >> 6, j = idx & 63;
  float v = x[n * 2] * W[j] + x[n * 2 + 1] * W[64 + j] + b[j];
  nf[idx] = fmaxf(v, 0.0f);
}

// h1 = relu(ea(E,5) @ W1(5,128) + b1)
__global__ void k_h1(const float* __restrict__ ea, const float* __restrict__ W1,
                     const float* __restrict__ b1, float* __restrict__ h1, int E) {
  int idx = blockIdx.x * blockDim.x + threadIdx.x;
  if (idx >= E * 128) return;
  int e = idx >> 7, j = idx & 127;
  float acc = b1[j];
#pragma unroll
  for (int i = 0; i < 5; ++i) acc += ea[e * 5 + i] * W1[i * 128 + j];
  h1[idx] = fmaxf(acc, 0.0f);
}

__global__ void k_deg(const int* __restrict__ ei, float* __restrict__ deg, int E) {
  int e = blockIdx.x * blockDim.x + threadIdx.x;
  if (e < E) atomicAdd(&deg[ei[E + e]], 1.0f);
}

// per edge: msg = nf[src] (1x64) @ ew_e (64x64, bf16); scatter-add to agg[dst].
// Lane owns outputs {2*lane, 2*lane+1}: one packed b32 load per K row; the
// wave-uniform nf row is read as 16 float4 loads.
__global__ __launch_bounds__(256)
void k_msg(const float* __restrict__ nf, const __bf16* __restrict__ ew,
           const int* __restrict__ ei, float* __restrict__ agg, int E) {
  int e = blockIdx.x * 8 + (threadIdx.x >> 5);
  if (e >= E) return;
  int lane = threadIdx.x & 31;
  int src = ei[e], dst = ei[E + e];
  const v2bf*   m2  = (const v2bf*)(ew + (size_t)e * 4096);
  const float4* nr4 = (const float4*)(nf + (size_t)src * 64);
  float a0 = 0.0f, a1 = 0.0f;
#pragma unroll 4
  for (int i4 = 0; i4 < 16; ++i4) {
    float4 nv = nr4[i4];
    v2bf p0 = m2[(i4 * 4 + 0) * 32 + lane];
    v2bf p1 = m2[(i4 * 4 + 1) * 32 + lane];
    v2bf p2 = m2[(i4 * 4 + 2) * 32 + lane];
    v2bf p3 = m2[(i4 * 4 + 3) * 32 + lane];
    a0 += nv.x * (float)p0[0]; a1 += nv.x * (float)p0[1];
    a0 += nv.y * (float)p1[0]; a1 += nv.y * (float)p1[1];
    a0 += nv.z * (float)p2[0]; a1 += nv.z * (float)p2[1];
    a0 += nv.w * (float)p3[0]; a1 += nv.w * (float)p3[1];
  }
  atomicAdd(&agg[dst * 64 + 2 * lane],     a0);
  atomicAdd(&agg[dst * 64 + 2 * lane + 1], a1);
}

// m = relu(agg/deg + (nf@root + conv_b))   (second term already in mR)
__global__ void k_mcomb(const float* __restrict__ agg, const float* __restrict__ deg,
                        float* __restrict__ mR, int N) {
  int idx = blockIdx.x * blockDim.x + threadIdx.x;
  if (idx >= N * 64) return;
  float d = fmaxf(deg[idx >> 6], 1.0f);
  mR[idx] = fmaxf(agg[idx] * fast_rcp(d) + mR[idx], 0.0f);
}

// GRU cell: h = (1-z)*n + z*h    (gi = m@Wi+bi (N,192), gh = h@Wh+bh)
__global__ void k_gru(const float* __restrict__ gi, const float* __restrict__ gh,
                      float* __restrict__ h, int N) {
  int idx = blockIdx.x * blockDim.x + threadIdx.x;
  if (idx >= N * 64) return;
  int n = idx >> 6, j = idx & 63;
  float ir = gi[n * 192 + j],       hr = gh[n * 192 + j];
  float iz = gi[n * 192 + 64 + j],  hz = gh[n * 192 + 64 + j];
  float in = gi[n * 192 + 128 + j], hn = gh[n * 192 + 128 + j];
  float r  = sigm(ir + hr);
  float z  = sigm(iz + hz);
  float nn = fast_tanh(in + r * hn);
  h[idx] = (1.0f - z) * nn + z * h[idx];
}

// LSTM cell (gA = q_star@Wi+bi, gB = hh@Wh+bh, each (G,256))
__global__ void k_lstm(const float* __restrict__ gA, const float* __restrict__ gB,
                       float* __restrict__ hh, float* __restrict__ cc, int G) {
  int idx = blockIdx.x * blockDim.x + threadIdx.x;
  if (idx >= G * 64) return;
  int g = idx >> 6, j = idx & 63;
  float gi = gA[g * 256 + j]       + gB[g * 256 + j];
  float gf = gA[g * 256 + 64 + j]  + gB[g * 256 + 64 + j];
  float gg = gA[g * 256 + 128 + j] + gB[g * 256 + 128 + j];
  float go = gA[g * 256 + 192 + j] + gB[g * 256 + 192 + j];
  float c2 = sigm(gf) * cc[idx] + sigm(gi) * fast_tanh(gg);
  cc[idx] = c2;
  hh[idx] = sigm(go) * fast_tanh(c2);
}

// Set2Set attention per graph (16 nodes): softmax(nf . hh), r = sum a*nf,
// q_star = [hh, r]. One 64-thread block per graph.
__global__ __launch_bounds__(64)
void k_attn(const float* __restrict__ nf, const float* __restrict__ hh,
            float* __restrict__ qstar) {
  __shared__ float se[16];
  __shared__ float sa[16];
  __shared__ float sinv;
  int g = blockIdx.x, t = threadIdx.x;
  if (t < 16) {
    const float* nrow = nf + ((size_t)g * 16 + t) * 64;
    const float* hr = hh + g * 64;
    float acc = 0.0f;
#pragma unroll 8
    for (int d = 0; d < 64; ++d) acc += nrow[d] * hr[d];
    se[t] = acc;
  }
  __syncthreads();
  if (t == 0) {
    float mx = se[0];
    for (int i = 1; i < 16; ++i) mx = fmaxf(mx, se[i]);
    float s = 0.0f;
    for (int i = 0; i < 16; ++i) { sa[i] = __expf(se[i] - mx); s += sa[i]; }
    sinv = fast_rcp(s);
  }
  __syncthreads();
  float inv = sinv;
  float r = 0.0f;
#pragma unroll
  for (int n = 0; n < 16; ++n)
    r += sa[n] * inv * nf[((size_t)g * 16 + n) * 64 + t];
  qstar[g * 128 + 64 + t] = r;
  qstar[g * 128 + t] = hh[g * 64 + t];
}

__global__ void k_gather_topo(const float* __restrict__ tbl, const int* __restrict__ topo,
                              float* __restrict__ topoE) {
  int idx = blockIdx.x * blockDim.x + threadIdx.x;
  if (idx >= BB * 128) return;
  int b = idx >> 7, j = idx & 127;
  topoE[idx] = tbl[(size_t)topo[b] * 128 + j];
}

// emb[b,p,o] = relu(bn(sum_l w[b,(p*128+o)*128+l] * qs[p*B+b, l]))
__global__ void k_block_einsum(const __bf16* __restrict__ w, const float* __restrict__ qs,
                               const float* __restrict__ bng, const float* __restrict__ bnb,
                               float* __restrict__ xcat, int xoff) {
  int idx = blockIdx.x * blockDim.x + threadIdx.x;
  if (idx >= BB * 384) return;
  int b = idx / 384, rem = idx - b * 384;
  int p = rem >> 7;
  const v8bf* wrow = (const v8bf*)(w + (size_t)b * 49152 + (size_t)rem * 128);
  const float* q   = qs + ((size_t)p * BB + b) * 128;
  float acc = 0.0f;
#pragma unroll
  for (int l8 = 0; l8 < 16; ++l8) {
    v8bf pk = wrow[l8];
    const float* qq = q + l8 * 8;
#pragma unroll
    for (int j = 0; j < 8; ++j) acc += (float)pk[j] * qq[j];
  }
  xcat[(size_t)b * 768 + xoff + rem] = fmaxf(acc * bng[rem] * BN_RSQ + bnb[rem], 0.0f);
}

// xint[b,o] = relu(bn(sum_f iw[b, f*64+o] * xcat[b,f]))
__global__ void k_int_einsum(const __bf16* __restrict__ iw, const float* __restrict__ xcat,
                             const float* __restrict__ bng, const float* __restrict__ bnb,
                             float* __restrict__ xint) {
  int idx = blockIdx.x * blockDim.x + threadIdx.x;
  if (idx >= BB * 64) return;
  int b = idx >> 6, o = idx & 63;
  const __bf16* w = iw + (size_t)b * 49152;
  const float*  xc = xcat + (size_t)b * 768;
  float acc = 0.0f;
  for (int f = 0; f < 768; ++f) acc += (float)w[f * 64 + o] * xc[f];
  xint[idx] = fmaxf(acc * bng[o] * BN_RSQ + bnb[o], 0.0f);
}

// head: relu(bn(x@pred_W)) @ out_W + out_b
__global__ void k_head(const float* __restrict__ xint, const float* __restrict__ predW,
                       const float* __restrict__ bnpg, const float* __restrict__ bnpb,
                       const float* __restrict__ outW, const float* __restrict__ outb,
                       float* __restrict__ out) {
  int b = blockIdx.x * blockDim.x + threadIdx.x;
  if (b >= BB) return;
  float acc = outb[0];
  for (int o2 = 0; o2 < 32; ++o2) {
    float s = 0.0f;
#pragma unroll 8
    for (int o = 0; o < 64; ++o) s += xint[b * 64 + o] * predW[o * 32 + o2];
    s = fmaxf(s * bnpg[o2] * BN_RSQ + bnpb[o2], 0.0f);
    acc += s * outW[o2];
  }
  out[b] = acc;
}

// ------------------------------- host side ---------------------------------
static inline void gemm(hipStream_t st, const float* A, const float* B,
                        const float* bias, void* C, int M, int N, int K,
                        int act, bool obf) {
  dim3 grid((N + 127) / 128, (M + 127) / 128), blk(256);
  if (obf) {
    if (act == 2) gemm_k<2, true><<<grid, blk, 0, st>>>(A, B, bias, C, N, K);
    else          gemm_k<0, true><<<grid, blk, 0, st>>>(A, B, bias, C, N, K);
  } else {
    gemm_k<0, false><<<grid, blk, 0, st>>>(A, B, bias, C, N, K);
  }
}

static inline void zero_f32(hipStream_t st, float* p, long n) {
  k_zero<<<(unsigned)((n + 255) / 256), 256, 0, st>>>(p, n);
}

// d_in flatten order (params dict recursively, then data args):
// [0..15] ngnn, [16..31] egnn (same order)
enum {
  P_EMB_W = 0, P_EMB_B, P_EW1, P_EB1, P_EW2, P_EB2, P_ROOT, P_CONVB,
  P_GRU_WI, P_GRU_WH, P_GRU_BI, P_GRU_BH, P_LSTM_WI, P_LSTM_WH, P_LSTM_BI, P_LSTM_BH
};
enum {
  IDX_TOPO_EMB = 32, IDX_NW_W, IDX_NW_B, IDX_EW_W, IDX_EW_B, IDX_IW_W, IDX_IW_B,
  IDX_BN_NODE_G, IDX_BN_NODE_B, IDX_BN_EDGE_G, IDX_BN_EDGE_B,
  IDX_BN_INT_G, IDX_BN_INT_B, IDX_PRED_W, IDX_BN_PRED_G, IDX_BN_PRED_B,
  IDX_OUT_W, IDX_OUT_B,
  IDX_NODE_X, IDX_NODE_EA, IDX_EDGE_X, IDX_EDGE_EA, IDX_TOPO,
  IDX_NODE_EI, IDX_NODE_BATCH, IDX_EDGE_EI, IDX_EDGE_BATCH
};

struct Bufs {
  __bf16* big;     // E*4096 bf16 slab: ew, later w/iw hypernet outputs
  float *h, *h1, *agg, *mR, *gi, *gh, *deg;
  float *hh, *cc, *gA, *gB;
  float *qsN, *qsE, *topoE, *xcat, *xint;
};

static void run_gnn(hipStream_t st, void* const* d_in, int pb,
                    const float* x, const float* ea, const int* ei,
                    const Bufs& w, float* qstar) {
  auto PF = [&](int i) { return (const float*)d_in[pb + i]; };
  const int N = NNODE, E = NEDGE;

  // node embedding + edge hidden
  k_embed<<<(N * 64 + 255) / 256, 256, 0, st>>>(x, PF(P_EMB_W), PF(P_EMB_B), w.h, N);
  k_h1<<<(E * 128 + 255) / 256, 256, 0, st>>>(ea, PF(P_EW1), PF(P_EB1), w.h1, E);

  // ew = h1 @ e_W2 + b2  (E,128)x(128,4096) -> bf16 (WMMA)
  gemm(st, w.h1, PF(P_EW2), PF(P_EB2), w.big, E, 4096, 128, 0, true);

  // degrees
  zero_f32(st, w.deg, N);
  k_deg<<<(E + 255) / 256, 256, 0, st>>>(ei, w.deg, E);

  // 3 message-passing + GRU steps (w.h doubles as nf and h)
  for (int d = 0; d < 3; ++d) {
    zero_f32(st, w.agg, (long)N * 64);
    k_msg<<<(E + 7) / 8, 256, 0, st>>>(w.h, w.big, ei, w.agg, E);
    gemm(st, w.h, PF(P_ROOT), PF(P_CONVB), w.mR, N, 64, 64, 0, false);   // WMMA
    k_mcomb<<<(N * 64 + 255) / 256, 256, 0, st>>>(w.agg, w.deg, w.mR, N);
    gemm(st, w.mR, PF(P_GRU_WI), PF(P_GRU_BI), w.gi, N, 192, 64, 0, false);
    gemm(st, w.h,  PF(P_GRU_WH), PF(P_GRU_BH), w.gh, N, 192, 64, 0, false);
    k_gru<<<(N * 64 + 255) / 256, 256, 0, st>>>(w.gi, w.gh, w.h, N);
  }

  // Set2Set readout (qstar also serves as q_star working buffer)
  zero_f32(st, qstar, (long)GG * 128);
  zero_f32(st, w.hh, (long)GG * 64);
  zero_f32(st, w.cc, (long)GG * 64);
  for (int s = 0; s < 3; ++s) {
    gemm(st, qstar, PF(P_LSTM_WI), PF(P_LSTM_BI), w.gA, GG, 256, 128, 0, false);
    gemm(st, w.hh,  PF(P_LSTM_WH), PF(P_LSTM_BH), w.gB, GG, 256, 64, 0, false);
    k_lstm<<<(GG * 64 + 255) / 256, 256, 0, st>>>(w.gA, w.gB, w.hh, w.cc, GG);
    k_attn<<<GG, 64, 0, st>>>(w.h, w.hh, qstar);
  }
}

extern "C" void kernel_launch(void* const* d_in, const int* in_sizes, int n_in,
                              void* d_out, int out_size, void* d_ws, size_t ws_size,
                              hipStream_t stream) {
  (void)in_sizes; (void)n_in; (void)out_size; (void)ws_size;
  auto F = [&](int i) { return (const float*)d_in[i]; };
  auto I = [&](int i) { return (const int*)d_in[i]; };

  // workspace carving
  char* ws = (char*)d_ws;
  size_t off = 0;
  auto carve = [&](size_t bytes) -> void* {
    void* p = ws + off;
    off += (bytes + 255) & ~(size_t)255;
    return p;
  };
  Bufs w;
  w.big   = (__bf16*)carve((size_t)NEDGE * 4096 * sizeof(__bf16));  // 201 MB slab
  w.h     = (float*)carve((size_t)NNODE * 64 * 4);
  w.h1    = (float*)carve((size_t)NEDGE * 128 * 4);
  w.agg   = (float*)carve((size_t)NNODE * 64 * 4);
  w.mR    = (float*)carve((size_t)NNODE * 64 * 4);
  w.gi    = (float*)carve((size_t)NNODE * 192 * 4);
  w.gh    = (float*)carve((size_t)NNODE * 192 * 4);
  w.deg   = (float*)carve((size_t)NNODE * 4);
  w.hh    = (float*)carve((size_t)GG * 64 * 4);
  w.cc    = (float*)carve((size_t)GG * 64 * 4);
  w.gA    = (float*)carve((size_t)GG * 256 * 4);
  w.gB    = (float*)carve((size_t)GG * 256 * 4);
  w.qsN   = (float*)carve((size_t)GG * 128 * 4);
  w.qsE   = (float*)carve((size_t)GG * 128 * 4);
  w.topoE = (float*)carve((size_t)BB * 128 * 4);
  w.xcat  = (float*)carve((size_t)BB * 768 * 4);
  w.xint  = (float*)carve((size_t)BB * 64 * 4);

  // topo embedding gather
  k_gather_topo<<<(BB * 128 + 255) / 256, 256, 0, stream>>>(
      F(IDX_TOPO_EMB), I(IDX_TOPO), w.topoE);

  // two GNNs (sequential; share the big slab)
  run_gnn(stream, d_in, 0,  F(IDX_NODE_X), F(IDX_NODE_EA), I(IDX_NODE_EI), w, w.qsN);
  run_gnn(stream, d_in, 16, F(IDX_EDGE_X), F(IDX_EDGE_EA), I(IDX_EDGE_EI), w, w.qsE);

  // hypernetwork (topoE @ W, tanh, bf16) + per-MOF einsum + BN/ReLU
  gemm(stream, w.topoE, F(IDX_NW_W), F(IDX_NW_B), w.big, BB, 49152, 128, 2, true);
  k_block_einsum<<<(BB * 384 + 255) / 256, 256, 0, stream>>>(
      w.big, w.qsN, F(IDX_BN_NODE_G), F(IDX_BN_NODE_B), w.xcat, 0);
  gemm(stream, w.topoE, F(IDX_EW_W), F(IDX_EW_B), w.big, BB, 49152, 128, 2, true);
  k_block_einsum<<<(BB * 384 + 255) / 256, 256, 0, stream>>>(
      w.big, w.qsE, F(IDX_BN_EDGE_G), F(IDX_BN_EDGE_B), w.xcat, 384);
  gemm(stream, w.topoE, F(IDX_IW_W), F(IDX_IW_B), w.big, BB, 49152, 128, 2, true);
  k_int_einsum<<<(BB * 64 + 255) / 256, 256, 0, stream>>>(
      w.big, w.xcat, F(IDX_BN_INT_G), F(IDX_BN_INT_B), w.xint);

  // prediction head
  k_head<<<(BB + 255) / 256, 256, 0, stream>>>(
      w.xint, F(IDX_PRED_W), F(IDX_BN_PRED_G), F(IDX_BN_PRED_B),
      F(IDX_OUT_W), F(IDX_OUT_B), (float*)d_out);
}
